// AlignNet_54623394071233
// MI455X (gfx1250) — matrix operand order repaired and checked by reading, not compiled, not run on hardware
//
#include <hip/hip_runtime.h>

// ---------------------------------------------------------------------------
// AlignNet forward for MI455X (gfx1250, wave32).
//
// All conv1d layers and correlation einsums run on v_wmma_f32_16x16x32_f16.
// This revision stores activations as f16, position-major [L][Cpad]
// (Cpad = ceil32(C)) and weights as f16 [K][CoutPad][CinPad] (prepped once
// per layer), so WMMA fragments are loaded as aligned vector f16 loads:
//   * B-fragment  = one 32-byte contiguous load (16 consecutive channels),
//   * A-fragment  = two v8h loads + one shuffle (contiguous element->K mapping:
//                   k = e + 16g),
//   * D-tile      = packed f16 v8h store (8 consecutive output channels).
// Channel padding is zero-filled -> no channel masks at all. Out-of-range
// positions redirect to a per-buffer zero row (reads) / dump row (writes);
// workspace is zeroed once per call with hipMemsetAsync so padding is never
// NaN. EXEC stays all-ones through every WMMA.
//
// Assumed d_in order: insertion-order DFS of setup_inputs() dict.
// d_out: concat of [64,512],[64,170],[64,85],[64,43] fp32.
// ---------------------------------------------------------------------------

typedef _Float16 half_t;
typedef __attribute__((ext_vector_type(16))) _Float16 v16h;
typedef __attribute__((ext_vector_type(8)))  _Float16 v8h;
typedef __attribute__((ext_vector_type(8)))  float    v8f;

__device__ __constant__ int c_att_map[20] = {0,0,0,1,1,1,2,2,2,2,3,3,3,3,4,4,4,4,0,1};

__device__ __forceinline__ v8f wmma16x16x32(const v16h& a, const v16h& b, const v8f& c) {
    return __builtin_amdgcn_wmma_f32_16x16x32_f16(false, a, false, b, (short)0, c,
                                                  false, false);
}
#define SHUF16(lo, hi) __builtin_shufflevector(lo, hi, 0,1,2,3,4,5,6,7,8,9,10,11,12,13,14,15)

// ---------------------------------------------------------------------------
// conv1d (+bias, optional relu). Input f16 [Lin+2][Cstride] (row Lin.. = zero
// row zrow, zrow+1 = dump). Weights f16 [K][CoutP][CinP] zero-padded. Output
// f16 [Lout+2][outStride] at channel offset oc. One wave = 32x32 outputs as
// 2x2 WMMA tiles; all fragment loads are aligned vector loads, no masks.
// ---------------------------------------------------------------------------
template <int K, int RELU>
__global__ void conv1d_wmma_kernel(
    const half_t* __restrict__ in, int inB, int Cstride, int Lin, int zrow,
    const half_t* __restrict__ wgtT, int CinP, int CoutP,
    const float* __restrict__ biasP,
    half_t* __restrict__ out, int outB, int outStride, int Lout, int dumpRow,
    int oc, int Cout, int stride, int pad, int dil)
{
    const int lane = threadIdx.x & 31;
    const int g    = lane >> 4;
    const int ln   = lane & 15;
    const int n0   = blockIdx.x * 32;
    const int c0   = blockIdx.y * 32;
    const half_t* inb  = in  + blockIdx.z * inB;
    half_t*       outb = out + blockIdx.z * outB;

    const int mA0 = c0 + ln, mA1 = mA0 + 16;       // A rows (out channels)
    const int nB0 = n0 + ln, nB1 = nB0 + 16;       // B cols (out positions)
    const int li0b = nB0 * stride - pad;
    const int li1b = nB1 * stride - pad;

    v8f acc00 = {}, acc01 = {}, acc10 = {}, acc11 = {};

#pragma unroll
    for (int kp = 0; kp < K; ++kp) {
        const int li0 = li0b + kp * dil;
        const int li1 = li1b + kp * dil;
        const int r0  = (li0 >= 0 && li0 < Lin) ? li0 : zrow;   // zero row
        const int r1  = (li1 >= 0 && li1 < Lin) ? li1 : zrow;
        const half_t* brow0 = inb + r0 * Cstride;
        const half_t* brow1 = inb + r1 * Cstride;
        const half_t* arow0 = wgtT + (kp * CoutP + mA0) * CinP;
        const half_t* arow1 = wgtT + (kp * CoutP + mA1) * CinP;
        for (int cc = 0; cc < CinP; cc += 32) {
            const int cA = cc + 16 * g;
            const int cB = cc + 16 * g;
            v8h al0 = *(const v8h*)(arow0 + cA);
            v8h ah0 = *(const v8h*)(arow0 + cA + 8);
            v8h al1 = *(const v8h*)(arow1 + cA);
            v8h ah1 = *(const v8h*)(arow1 + cA + 8);
            v16h a0 = SHUF16(al0, ah0);
            v16h a1 = SHUF16(al1, ah1);
            v16h b0 = *(const v16h*)(brow0 + cB);
            v16h b1 = *(const v16h*)(brow1 + cB);
            acc00 = wmma16x16x32(a0, b0, acc00);
            acc01 = wmma16x16x32(a0, b1, acc01);
            acc10 = wmma16x16x32(a1, b0, acc10);
            acc11 = wmma16x16x32(a1, b1, acc11);
        }
    }

    auto store_tile = [&](const v8f& acc, int mbase, int nbase) {
        const int mrow = mbase + g * 8;            // 8 consecutive out channels
        const int no   = nbase + ln;
        const int row  = (no < Lout) ? no : dumpRow;
        const v8f bv = *(const v8f*)(biasP + mrow);
        v8h ph;
#pragma unroll
        for (int r = 0; r < 8; ++r) {
            float v = acc[r] + bv[r];
            if (RELU) v = fmaxf(v, 0.0f);
            ph[r] = (half_t)(((mrow + r) < Cout) ? v : 0.0f);
        }
        half_t* p = outb + row * outStride + oc + mrow;
        if ((oc & 7) == 0) {
            *(v8h*)p = ph;                         // aligned packed store
        } else {
#pragma unroll
            for (int r = 0; r < 8; ++r) p[r] = ph[r];
        }
    };
    store_tile(acc00, c0,      n0);
    store_tile(acc01, c0,      n0 + 16);
    store_tile(acc10, c0 + 16, n0);
    store_tile(acc11, c0 + 16, n0 + 16);
}

// ---------------------------------------------------------------------------
// Correlation einsum('bcj,bcl->bjl'). A = warped f16 [La+2][aStride],
// B = vfeat f16 [Lv+2][vStride]; both zero-padded in channels. Output f16
// into feat channels [0, La) (La is a multiple of 8).
// ---------------------------------------------------------------------------
__global__ void corr_wmma_kernel(
    const half_t* __restrict__ A, int aB, int aStride, int La, int zrowA,
    const half_t* __restrict__ V, int vB, int vStride, int Lv, int zrowV,
    half_t* __restrict__ out, int outB, int outStride, int dumpRow, int CPad)
{
    const int lane = threadIdx.x & 31;
    const int g    = lane >> 4;
    const int ln   = lane & 15;
    const int n0   = blockIdx.x * 32;   // Lv tile
    const int j0   = blockIdx.y * 32;   // La tile
    const half_t* Ab   = A + blockIdx.z * aB;
    const half_t* Vb   = V + blockIdx.z * vB;
    half_t*       outb = out + blockIdx.z * outB;

    const int mA0 = j0 + ln, mA1 = mA0 + 16;
    const int nB0 = n0 + ln, nB1 = nB0 + 16;
    const int ra0 = (mA0 < La) ? mA0 : zrowA;
    const int ra1 = (mA1 < La) ? mA1 : zrowA;
    const int rb0 = (nB0 < Lv) ? nB0 : zrowV;
    const int rb1 = (nB1 < Lv) ? nB1 : zrowV;
    const half_t* arow0 = Ab + ra0 * aStride;
    const half_t* arow1 = Ab + ra1 * aStride;
    const half_t* brow0 = Vb + rb0 * vStride;
    const half_t* brow1 = Vb + rb1 * vStride;

    v8f acc00 = {}, acc01 = {}, acc10 = {}, acc11 = {};
    for (int cc = 0; cc < CPad; cc += 32) {
        const int cA = cc + 16 * g;
        const int cB = cc + 16 * g;
        v16h a0 = SHUF16(*(const v8h*)(arow0 + cA), *(const v8h*)(arow0 + cA + 8));
        v16h a1 = SHUF16(*(const v8h*)(arow1 + cA), *(const v8h*)(arow1 + cA + 8));
        v16h b0 = *(const v16h*)(brow0 + cB);
        v16h b1 = *(const v16h*)(brow1 + cB);
        acc00 = wmma16x16x32(a0, b0, acc00);
        acc01 = wmma16x16x32(a0, b1, acc01);
        acc10 = wmma16x16x32(a1, b0, acc10);
        acc11 = wmma16x16x32(a1, b1, acc11);
    }

    auto store_tile = [&](const v8f& acc, int mbase, int nbase) {
        const int mrow = mbase + g * 8;
        const int no   = nbase + ln;
        const int row  = ((no < Lv) & (mrow < La)) ? no : dumpRow;
        v8h ph;
#pragma unroll
        for (int r = 0; r < 8; ++r) ph[r] = (half_t)acc[r];
        *(v8h*)(outb + row * outStride + mrow) = ph;
    };
    store_tile(acc00, j0,      n0);
    store_tile(acc01, j0,      n0 + 16);
    store_tile(acc10, j0 + 16, n0);
    store_tile(acc11, j0 + 16, n0 + 16);
}

// ---------------------------------------------------------------------------
// Prep + glue kernels.
// ---------------------------------------------------------------------------

// weights [Cout][Cin][K] f32 -> wgtT [K][CoutP][CinP] f16 (zero-padded);
// bias -> biasP [CoutP] f32 (zero-padded)
__global__ void wprep_kernel(const float* __restrict__ w, const float* __restrict__ b,
                             half_t* __restrict__ wt, float* __restrict__ bp,
                             int Cout, int Cin, int K, int CoutP, int CinP, int n)
{
    int idx = blockIdx.x * blockDim.x + threadIdx.x;
    if (idx >= n) return;
    int c  = idx % CinP;
    int m  = (idx / CinP) % CoutP;
    int kp = idx / (CinP * CoutP);
    float v = (m < Cout && c < Cin) ? w[(m * Cin + c) * K + kp] : 0.0f;
    wt[idx] = (half_t)v;
    if (kp == 0 && c == 0) bp[m] = (m < Cout) ? b[m] : 0.0f;
}

// softmax(kp_att)[ATT_MAP] scale + transpose -> f16 [B][514][96]
__global__ void vf_prep_kernel(const float* __restrict__ vfeat,
                               const float* __restrict__ lg,
                               half_t* __restrict__ out, int n)
{
    int idx = blockIdx.x * blockDim.x + threadIdx.x;
    if (idx >= n) return;
    int c = idx % 96;
    int t = (idx / 96) % 512;
    int b = idx / (96 * 512);
    float a0 = lg[0], a1 = lg[1], a2 = lg[2], a3 = lg[3], a4 = lg[4];
    float mx = fmaxf(fmaxf(fmaxf(a0, a1), fmaxf(a2, a3)), a4);
    float e[5];
    e[0] = __expf(a0 - mx); e[1] = __expf(a1 - mx); e[2] = __expf(a2 - mx);
    e[3] = __expf(a3 - mx); e[4] = __expf(a4 - mx);
    float s = e[0] + e[1] + e[2] + e[3] + e[4];
    float val = 0.0f;
    if (c < 80) {
        int kp = c >> 2;
        val = vfeat[((b * 512 + t) * 20 + kp) * 4 + (c & 3)] * (e[c_att_map[kp]] / s);
    }
    out[b * (514 * 96) + t * 96 + c] = (half_t)val;
}

// audio f32 [B][80][2056] -> f16 [B][2058][96]
__global__ void audio_prep_kernel(const float* __restrict__ a,
                                  half_t* __restrict__ out, int n)
{
    int idx = blockIdx.x * blockDim.x + threadIdx.x;
    if (idx >= n) return;
    int c = idx % 96;
    int t = (idx / 96) % 2056;
    int b = idx / (96 * 2056);
    float v = (c < 80) ? a[(b * 80 + c) * 2056 + t] : 0.0f;
    out[b * (2058 * 96) + t * 96 + c] = (half_t)v;
}

// linear upsample fp32->fp32 (C=1), align_corners=False
__global__ void upsample_kernel(const float* __restrict__ in, int inB, int L,
                                float* __restrict__ out, int outB, int Lo, int n)
{
    int idx = blockIdx.x * blockDim.x + threadIdx.x;
    if (idx >= n) return;
    int o = idx % Lo;
    int b = idx / Lo;
    float pos = ((float)o + 0.5f) * ((float)L / (float)Lo) - 0.5f;
    pos = fminf(fmaxf(pos, 0.0f), (float)(L - 1));
    int lo = (int)floorf(pos);
    int hi = min(lo + 1, L - 1);
    float w = pos - (float)lo;
    const float* p = in + b * inB;
    out[b * outB + o] = p[lo] * (1.0f - w) + p[hi] * w;
}

// linear upsample fp32 -> f16 channel `oc` of a [Lo][outStride] f16 buffer
__global__ void upsample_feat_kernel(const float* __restrict__ in, int inB, int L,
                                     half_t* __restrict__ out, int outB,
                                     int outStride, int oc, int Lo, int n)
{
    int idx = blockIdx.x * blockDim.x + threadIdx.x;
    if (idx >= n) return;
    int o = idx % Lo;
    int b = idx / Lo;
    float pos = ((float)o + 0.5f) * ((float)L / (float)Lo) - 0.5f;
    pos = fminf(fmaxf(pos, 0.0f), (float)(L - 1));
    int lo = (int)floorf(pos);
    int hi = min(lo + 1, L - 1);
    float w = pos - (float)lo;
    const float* p = in + b * inB;
    out[b * outB + o * outStride + oc] = (half_t)(p[lo] * (1.0f - w) + p[hi] * w);
}

// grid_sample bilinear (align_corners=True, zeros) on f16 [L+2][Cpad] rows
__global__ void warp_kernel(const half_t* __restrict__ x, int xB, int Cpad, int Cv,
                            int L, const float* __restrict__ ind, int indB,
                            half_t* __restrict__ out, int outB, int n)
{
    int idx = blockIdx.x * blockDim.x + threadIdx.x;
    if (idx >= n) return;
    int c = idx % Cpad;
    int o = (idx / Cpad) % L;
    int b = idx / (Cpad * L);
    float ix = (ind[b * indB + o] + 1.0f) * 0.5f * (float)(L - 1);
    float x0 = floorf(ix);
    float w  = ix - x0;
    int i0 = (int)x0;
    int i1 = i0 + 1;
    const half_t* p = x + b * xB;
    float v0 = (i0 >= 0 && i0 < L) ? (float)p[i0 * Cpad + c] : 0.0f;
    float v1 = (i1 >= 0 && i1 < L) ? (float)p[i1 * Cpad + c] : 0.0f;
    float val = (c < Cv) ? (v0 * (1.0f - w) + v1 * w) : 0.0f;
    out[b * outB + o * Cpad + c] = (half_t)val;
}

// copy v_feats channels into feat buffer at channel offset oc
__global__ void vcopy_kernel(const half_t* __restrict__ in, int inB, int inStride,
                             half_t* __restrict__ out, int outB, int outStride,
                             int oc, int Cv, int Lv, int n)
{
    int idx = blockIdx.x * blockDim.x + threadIdx.x;
    if (idx >= n) return;
    int c = idx % Cv;
    int l = (idx / Cv) % Lv;
    int b = idx / (Cv * Lv);
    out[b * outB + l * outStride + oc + c] = in[b * inB + l * inStride + c];
}

// channel-0 of f16 [Lv+2][stride] -> fp32 flow [B][Lv]
__global__ void flow_extract_kernel(const half_t* __restrict__ in, int inB,
                                    int inStride, float* __restrict__ out,
                                    int Lv, int n)
{
    int idx = blockIdx.x * blockDim.x + threadIdx.x;
    if (idx >= n) return;
    int l = idx % Lv;
    int b = idx / Lv;
    out[idx] = (float)in[b * inB + l * inStride];
}

// flow += channel-0 of f16 dp output
__global__ void addf16_kernel(float* __restrict__ a, const half_t* __restrict__ d,
                              int dB, int dStride, int Lv, int n)
{
    int idx = blockIdx.x * blockDim.x + threadIdx.x;
    if (idx >= n) return;
    int l = idx % Lv;
    int b = idx / Lv;
    a[idx] += (float)d[b * dB + l * dStride];
}

// ---------------------------------------------------------------------------
// Host orchestration
// ---------------------------------------------------------------------------
struct KSP { int k, s, p; };
struct HBuf { half_t* p; int stride; int zrow; int bstride; };

extern "C" void kernel_launch(void* const* d_in, const int* in_sizes, int n_in,
                              void* d_out, int out_size, void* d_ws, size_t ws_size,
                              hipStream_t stream)
{
    (void)in_sizes; (void)n_in; (void)out_size; (void)ws_size;
    const int Bc = 64;
    auto ceil32 = [](int x) { return (x + 31) & ~31; };

    // ---- unpack inputs (insertion-order DFS, w before b) ----
    int idx = 0;
    const float* video_feature = (const float*)d_in[idx++];
    const float* audio_feature = (const float*)d_in[idx++];
    const float* kp_att        = (const float*)d_in[idx++];
    const int nlayers[4] = {5, 3, 3, 3};
    const float *vw[4][5], *vb[4][5], *aw[4][5], *ab[4][5];
    for (int i = 0; i < 4; ++i)
        for (int j = 0; j < nlayers[i]; ++j) {
            vw[i][j] = (const float*)d_in[idx++]; vb[i][j] = (const float*)d_in[idx++];
        }
    for (int i = 0; i < 4; ++i)
        for (int j = 0; j < nlayers[i]; ++j) {
            aw[i][j] = (const float*)d_in[idx++]; ab[i][j] = (const float*)d_in[idx++];
        }
    const float *ew[4][2], *eb[4][2];
    for (int i = 0; i < 4; ++i)
        for (int j = 0; j < 2; ++j) {
            ew[i][j] = (const float*)d_in[idx++]; eb[i][j] = (const float*)d_in[idx++];
        }
    const float *pw[4], *pb[4];
    for (int i = 0; i < 4; ++i) { pw[i] = (const float*)d_in[idx++]; pb[i] = (const float*)d_in[idx++]; }
    const float *dw[4], *db[4];
    for (int j = 0; j < 4; ++j) { dw[j] = (const float*)d_in[idx++]; db[j] = (const float*)d_in[idx++]; }

    // ---- geometry ----
    const int Cv[4]   = {128, 64, 32, 16};
    const int CvP[4]  = {128, 64, 32, 32};
    const int LvE[4]  = {170, 85, 43, 22};
    const int LaE[4]  = {64, 32, 16, 8};
    const int cinx[4] = {193, 97, 49, 24};
    const int cadd[4] = {32, 16, 8, 4};
    const int Ctot[4] = {257, 129, 65, 32};
    const int CtP[4]  = {288, 160, 96, 32};
    const int secL[4] = {512, 170, 85, 43};
    const int sec[4]  = {0, Bc * 512, Bc * (512 + 170), Bc * (512 + 170 + 85)};

    // ---- workspace bump allocator ----
    char* base = (char*)d_ws;
    size_t off = 0;
    auto allocB = [&](size_t bytes) {
        void* p = base + off;
        off = (off + bytes + 255) & ~(size_t)255;
        return p;
    };
    auto hbuf = [&](int L, int cpad) {
        HBuf b; b.stride = cpad; b.zrow = L; b.bstride = (L + 2) * cpad;
        b.p = (half_t*)allocB((size_t)Bc * b.bstride * sizeof(half_t));
        return b;
    };

    HBuf vf16  = hbuf(512, 96);
    HBuf aud16 = hbuf(2056, 96);
    HBuf ping  = hbuf(515, 128);
    HBuf pong  = hbuf(515, 128);
    HBuf vfeatB[4], afeatB[4], featB[4], predB[4];
    for (int i = 0; i < 4; ++i) vfeatB[i] = hbuf(LvE[i], CvP[i]);
    for (int i = 0; i < 4; ++i) afeatB[i] = hbuf(LaE[i], CvP[i]);
    for (int i = 0; i < 4; ++i) featB[i]  = hbuf(LvE[i], CtP[i]);
    for (int i = 0; i < 4; ++i) predB[i]  = hbuf(LvE[i], 32);
    HBuf warpB[3];
    for (int i = 0; i < 3; ++i) warpB[i] = hbuf(LaE[i], CvP[i]);
    HBuf dpO[4] = {hbuf(170, 64), hbuf(170, 32), hbuf(170, 32), hbuf(170, 32)};
    float* flw[4];
    for (int i = 0; i < 4; ++i) flw[i] = (float*)allocB((size_t)Bc * LvE[i] * 4);
    float* ufa = (float*)allocB((size_t)Bc * 64 * 4);

    // ---- weight registry (fixed order -> fixed addresses per call) ----
    struct WSpec { const float *w, *b; int Cout, Cin, K; half_t* wt; float* bp; };
    WSpec W[44]; int nw = 0;
    auto addw = [&](const float* w, const float* b, int Cout, int Cin, int K) {
        W[nw] = {w, b, Cout, Cin, K, nullptr, nullptr}; ++nw;
    };
    const KSP vspec[4][5] = {
        {{5,3,1},{3,1,1},{3,1,1},{3,1,1},{3,1,1}},
        {{3,2,1},{3,1,1},{3,1,1},{0,0,0},{0,0,0}},
        {{3,2,1},{3,1,1},{3,1,1},{0,0,0},{0,0,0}},
        {{3,2,1},{3,1,1},{3,1,1},{0,0,0},{0,0,0}}};
    const KSP aspec[4][5] = {
        {{7,4,4},{5,2,1},{5,2,1},{3,2,1},{3,1,1}},
        {{3,2,1},{3,1,1},{3,1,1},{0,0,0},{0,0,0}},
        {{3,2,1},{3,1,1},{3,1,1},{0,0,0},{0,0,0}},
        {{3,2,1},{3,1,1},{3,1,1},{0,0,0},{0,0,0}}};
    {
        int Cc = 80;
        for (int i = 0; i < 4; ++i) { int Co = 128 >> i;
            for (int j = 0; j < nlayers[i]; ++j) { addw(vw[i][j], vb[i][j], Co, Cc, vspec[i][j].k); Cc = Co; } }
        Cc = 80;
        for (int i = 0; i < 4; ++i) { int Co = 128 >> i;
            for (int j = 0; j < nlayers[i]; ++j) { addw(aw[i][j], ab[i][j], Co, Cc, aspec[i][j].k); Cc = Co; } }
        for (int i = 0; i < 4; ++i) {
            addw(ew[i][0], eb[i][0], cadd[i], cinx[i], 3);
            addw(ew[i][1], eb[i][1], cadd[i], cinx[i] + cadd[i], 3);
        }
        for (int i = 0; i < 4; ++i) addw(pw[i], pb[i], 1, Ctot[i], 3);
        const int dci[4] = {257, 64, 16, 4}, dco[4] = {64, 16, 4, 1};
        for (int j = 0; j < 4; ++j) addw(dw[j], db[j], dco[j], dci[j], 3);
    }
    for (int i = 0; i < nw; ++i) {
        int CoP = ceil32(W[i].Cout), CiP = ceil32(W[i].Cin);
        W[i].wt = (half_t*)allocB((size_t)W[i].K * CoP * CiP * sizeof(half_t));
        W[i].bp = (float*)allocB((size_t)CoP * 4);
    }
    const int voff[4] = {0, 5, 8, 11};   // layer offsets inside a pyramid
    auto widx_video = [&](int i, int j) { return voff[i] + j; };
    auto widx_audio = [&](int i, int j) { return 14 + voff[i] + j; };
    auto widx_ext   = [&](int i, int j) { return 28 + i * 2 + j; };
    auto widx_pred  = [&](int i) { return 36 + i; };
    auto widx_dp    = [&](int j) { return 40 + j; };

    // ---- zero the whole workspace (padding rows/channels stay zero) ----
    hipMemsetAsync(d_ws, 0, off, stream);

    auto ew1 = [&](int n) { return dim3((n + 255) / 256); };

    // ---- weight prep ----
    for (int i = 0; i < nw; ++i) {
        int CoP = ceil32(W[i].Cout), CiP = ceil32(W[i].Cin);
        int n = W[i].K * CoP * CiP;
        wprep_kernel<<<ew1(n), 256, 0, stream>>>(W[i].w, W[i].b, W[i].wt, W[i].bp,
                                                 W[i].Cout, W[i].Cin, W[i].K, CoP, CiP, n);
    }

    // ---- conv launcher ----
    auto convL = [&](int wi, const HBuf& ib, int Lin, int Cin,
                     const HBuf& ob, int Lout, int Cout, int oc,
                     int K, int s, int p, int d, int relu) {
        int CiP = ceil32(Cin), CoP = ceil32(Cout);
        dim3 grid((Lout + 31) / 32, CoP / 32, Bc);
#define LC(KK, RR) conv1d_wmma_kernel<KK, RR><<<grid, dim3(32), 0, stream>>>(     \
            ib.p, ib.bstride, ib.stride, Lin, ib.zrow, W[wi].wt, CiP, CoP,        \
            W[wi].bp, ob.p, ob.bstride, ob.stride, Lout, ob.zrow + 1, oc, Cout,   \
            s, p, d)
        if (relu) { if (K == 3) LC(3, 1); else if (K == 5) LC(5, 1); else LC(7, 1); }
        else      { if (K == 3) LC(3, 0); else if (K == 5) LC(5, 0); else LC(7, 0); }
#undef LC
    };

    // ---- input prep ----
    vf_prep_kernel<<<ew1(Bc * 512 * 96), 256, 0, stream>>>(video_feature, kp_att,
                                                           vf16.p, Bc * 512 * 96);
    audio_prep_kernel<<<ew1(Bc * 2056 * 96), 256, 0, stream>>>(audio_feature, aud16.p,
                                                               Bc * 2056 * 96);

    // ---- pyramids ----
    auto run_pyr = [&](const HBuf& input, int L0c, const KSP spec[4][5],
                       HBuf dst[4], bool isVideo) {
        const HBuf* cur = &input;
        int Lc = L0c, Cc = 80;
        for (int i = 0; i < 4; ++i) {
            int Co = 128 >> i;
            for (int j = 0; j < nlayers[i]; ++j) {
                KSP sp = spec[i][j];
                int Lout = (Lc + 2 * sp.p - sp.k) / sp.s + 1;
                const HBuf* ob = (j == nlayers[i] - 1) ? &dst[i]
                                 : ((cur == &ping) ? &pong : &ping);
                int wi = isVideo ? widx_video(i, j) : widx_audio(i, j);
                convL(wi, *cur, Lc, Cc, *ob, Lout, Co, 0, sp.k, sp.s, sp.p, 1, 1);
                cur = ob; Lc = Lout; Cc = Co;
            }
        }
    };
    run_pyr(vf16, 512, vspec, vfeatB, true);
    run_pyr(aud16, 2056, aspec, afeatB, false);

    // ---- coarse-to-fine decoder ----
    float* out0 = (float*)d_out;
    for (int i = 3; i >= 0; --i) {
        const HBuf& fb = featB[i];
        if (i == 3) {
            dim3 grid((LvE[3] + 31) / 32, ceil32(LaE[3]) / 32, Bc);
            corr_wmma_kernel<<<grid, dim3(32), 0, stream>>>(
                afeatB[3].p, afeatB[3].bstride, afeatB[3].stride, LaE[3], afeatB[3].zrow,
                vfeatB[3].p, vfeatB[3].bstride, vfeatB[3].stride, LvE[3], vfeatB[3].zrow,
                fb.p, fb.bstride, fb.stride, fb.zrow + 1, CvP[3]);
        } else {
            int n = Bc * LaE[i];
            upsample_kernel<<<ew1(n), 256, 0, stream>>>(
                flw[i + 1], LvE[i + 1], LvE[i + 1], ufa, LaE[i], LaE[i], n);
            n = Bc * LaE[i] * CvP[i];
            warp_kernel<<<ew1(n), 256, 0, stream>>>(
                afeatB[i].p, afeatB[i].bstride, CvP[i], Cv[i], LaE[i],
                ufa, LaE[i], warpB[i].p, warpB[i].bstride, n);
            dim3 grid((LvE[i] + 31) / 32, ceil32(LaE[i]) / 32, Bc);
            corr_wmma_kernel<<<grid, dim3(32), 0, stream>>>(
                warpB[i].p, warpB[i].bstride, warpB[i].stride, LaE[i], warpB[i].zrow,
                vfeatB[i].p, vfeatB[i].bstride, vfeatB[i].stride, LvE[i], vfeatB[i].zrow,
                fb.p, fb.bstride, fb.stride, fb.zrow + 1, CvP[i]);
            n = Bc * LvE[i];
            upsample_feat_kernel<<<ew1(n), 256, 0, stream>>>(
                flw[i + 1], LvE[i + 1], LvE[i + 1],
                fb.p, fb.bstride, fb.stride, cinx[i] - 1, LvE[i], n);
            n = Bc * secL[i + 1];
            upsample_kernel<<<ew1(n), 256, 0, stream>>>(
                flw[i + 1], LvE[i + 1], LvE[i + 1],
                out0 + sec[i + 1], secL[i + 1], secL[i + 1], n);
        }
        int n = Bc * LvE[i] * Cv[i];
        vcopy_kernel<<<ew1(n), 256, 0, stream>>>(
            vfeatB[i].p, vfeatB[i].bstride, vfeatB[i].stride,
            fb.p, fb.bstride, fb.stride, LaE[i], Cv[i], LvE[i], n);
        convL(widx_ext(i, 0), fb, LvE[i], cinx[i], fb, LvE[i], cadd[i],
              cinx[i], 3, 1, 1, 1, 1);
        convL(widx_ext(i, 1), fb, LvE[i], cinx[i] + cadd[i], fb, LvE[i], cadd[i],
              cinx[i] + cadd[i], 3, 1, 1, 1, 1);
        convL(widx_pred(i), fb, LvE[i], Ctot[i], predB[i], LvE[i], 1, 0,
              3, 1, 1, 1, 0);
        n = Bc * LvE[i];
        flow_extract_kernel<<<ew1(n), 256, 0, stream>>>(
            predB[i].p, predB[i].bstride, predB[i].stride, flw[i], LvE[i], n);
    }

    // ---- dilated-conv refinement on level-0 feat ----
    {
        const int dci[4] = {257, 64, 16, 4}, dco[4] = {64, 16, 4, 1};
        const HBuf* cur = &featB[0];
        for (int j = 0; j < 4; ++j) {
            int dil = 1 << j;
            convL(widx_dp(j), *cur, 170, dci[j], dpO[j], 170, dco[j], 0,
                  3, 1, dil, dil, 1);
            cur = &dpO[j];
        }
        int n = Bc * 170;
        addf16_kernel<<<ew1(n), 256, 0, stream>>>(flw[0], dpO[3].p, dpO[3].bstride,
                                                  dpO[3].stride, 170, n);
        n = Bc * 512;
        upsample_kernel<<<ew1(n), 256, 0, stream>>>(
            flw[0], 170, 170, out0 + sec[0], 512, 512, n);
    }
}
